// EEG_GAT_65712999628910
// MI455X (gfx1250) — matrix operand order repaired and checked by reading, not compiled, not run on hardware
//
#include <hip/hip_runtime.h>
#include <hip/hip_bf16.h>
#include <stdint.h>

// ---------------- problem constants ----------------
static constexpr int Nn    = 65536;
static constexpr int Gn    = 1024;
static constexpr int NPG   = 64;
static constexpr int En    = 1048576;
static constexpr int INC   = 128;
static constexpr int HIDC  = 64;
static constexpr int HEADS = 4;
static constexpr int NCLS  = 5;

typedef __attribute__((ext_vector_type(8)))  _Float16 v8h_t;
typedef __attribute__((ext_vector_type(16))) _Float16 v16h_t;
typedef __attribute__((ext_vector_type(8)))  float    v8f_t;

// ---------------- helpers ----------------
__device__ __forceinline__ float lrelu02(float v) { return v > 0.f ? v : 0.2f * v; }
__device__ __forceinline__ unsigned f2ord(float f) {
    unsigned u = __float_as_uint(f);
    return (u & 0x80000000u) ? ~u : (u | 0x80000000u);
}
__device__ __forceinline__ float ord2f(unsigned u) {
    return (u & 0x80000000u) ? __uint_as_float(u & 0x7fffffffu) : __uint_as_float(~u);
}

// CDNA5 async global->LDS copy (ASYNCcnt-tracked, no VGPR staging).
// lds = low 32 bits of a generic LDS pointer (byte offset in workgroup LDS),
// ga  = 64-bit global address; each active lane moves 16 bytes.
__device__ __forceinline__ void async_ld_b128(unsigned lds, unsigned long long ga) {
    asm volatile("global_load_async_to_lds_b128 %0, %1, off"
                 :: "v"(lds), "v"(ga) : "memory");
}
__device__ __forceinline__ void wait_async0() {
    asm volatile("s_wait_asynccnt 0x0" ::: "memory");
}

// ---------------- generic kernels ----------------
__global__ void k_zero(float* p, int n) {
    int i = blockIdx.x * blockDim.x + threadIdx.x;
    if (i < n) p[i] = 0.f;
}

__global__ void k_f32_to_f16(const float* s, _Float16* d, int n) {
    int i = blockIdx.x * blockDim.x + threadIdx.x;
    if (i < n) d[i] = (_Float16)s[i];
}

// W[K,Co] (row-major f32) -> Wt[Co,K] (row-major f16)
__global__ void k_transpose_f16(const float* W, _Float16* Wt, int K, int Co) {
    int i = blockIdx.x * blockDim.x + threadIdx.x;
    if (i >= K * Co) return;
    int k = i / Co, n = i % Co;
    Wt[(size_t)n * K + k] = (_Float16)W[i];
}

// ---------------- WMMA GEMM: C[M,Co] = A[M,K] @ Wt^T (+bias) ----------------
// A: [M,K] f16 row-major.  Bt: [Co,K] f16 row-major (i.e. W transposed).
// block = 128 threads (4 wave32). block tile 64x64; wave w = rows 16w..16w+15,
// each wave computes 16x64 (4 accumulators, A-fragment reused 4x per k-step).
// K <= 256, K % 32 == 0, M % 64 == 0, Co % 64 == 0.
__global__ __launch_bounds__(128)
void k_gemm_wmma(const _Float16* __restrict__ A, const _Float16* __restrict__ Bt,
                 const float* __restrict__ bias, float* __restrict__ C,
                 int M, int K, int Co) {
    __shared__ _Float16 Al[64 * 256];
    __shared__ _Float16 Bl[64 * 256];
    const int m0 = blockIdx.x * 64;
    const int n0 = blockIdx.y * 64;
    const int t  = threadIdx.x;
    const int K8 = K >> 3;

    // async-stage both tiles straight into LDS (16B per lane per op)
    for (int i = t; i < 64 * K8; i += 128) {
        int r = i / K8, c = (i % K8) * 8;
        async_ld_b128((unsigned)(uintptr_t)&Al[r * K + c],
                      (unsigned long long)(uintptr_t)&A[(size_t)(m0 + r) * K + c]);
    }
    for (int i = t; i < 64 * K8; i += 128) {
        int r = i / K8, c = (i % K8) * 8;
        async_ld_b128((unsigned)(uintptr_t)&Bl[r * K + c],
                      (unsigned long long)(uintptr_t)&Bt[(size_t)(n0 + r) * K + c]);
    }
    wait_async0();
    __syncthreads();

    const int lane = t & 31;
    const int wid  = t >> 5;
    const int half = lane >> 4;   // 0: lanes 0-15, 1: lanes 16-31
    const int lr   = lane & 15;

    v8f_t acc0 = {0.f, 0.f, 0.f, 0.f, 0.f, 0.f, 0.f, 0.f};
    v8f_t acc1 = acc0, acc2 = acc0, acc3 = acc0;

    for (int k0 = 0; k0 < K; k0 += 32) {
        // A fragment for this wave's 16 rows (contiguous k = 16*half + s, matching B)
        const _Float16* ap = &Al[(size_t)(wid * 16 + lr) * K + k0 + half * 16];
        v8h_t alo = *(const v8h_t*)ap;
        v8h_t ahi = *(const v8h_t*)(ap + 8);
        v16h_t a  = __builtin_shufflevector(alo, ahi, 0, 1, 2, 3, 4, 5, 6, 7,
                                            8, 9, 10, 11, 12, 13, 14, 15);
        // 4 B fragments (N quadrants), A reused
        const _Float16* bp = &Bl[(size_t)lr * K + k0 + half * 16];
        v16h_t b0 = *(const v16h_t*)(bp + 0 * 16 * K);
        v16h_t b1 = *(const v16h_t*)(bp + 1 * 16 * K);
        v16h_t b2 = *(const v16h_t*)(bp + 2 * 16 * K);
        v16h_t b3 = *(const v16h_t*)(bp + 3 * 16 * K);
        acc0 = __builtin_amdgcn_wmma_f32_16x16x32_f16(false, a, false, b0, (short)0, acc0, false, false);
        acc1 = __builtin_amdgcn_wmma_f32_16x16x32_f16(false, a, false, b1, (short)0, acc1, false, false);
        acc2 = __builtin_amdgcn_wmma_f32_16x16x32_f16(false, a, false, b2, (short)0, acc2, false, false);
        acc3 = __builtin_amdgcn_wmma_f32_16x16x32_f16(false, a, false, b3, (short)0, acc3, false, false);
    }

    // C/D layout: vgpr r -> M=r (+8 for lanes 16-31), N = lane%16
    const int rbase = m0 + wid * 16 + half * 8;
#pragma unroll
    for (int nq = 0; nq < 4; nq++) {
        v8f_t acc = nq == 0 ? acc0 : nq == 1 ? acc1 : nq == 2 ? acc2 : acc3;
        int col = n0 + nq * 16 + lr;
        float bv = bias ? bias[col] : 0.f;
#pragma unroll
        for (int r = 0; r < 8; r++)
            C[(size_t)(rbase + r) * Co + col] = acc[r] + bv;
    }
}

// ---------------- GAT scalar kernels ----------------
__global__ void k_att_pair(const float* __restrict__ xs, const float* __restrict__ asrc,
                           const float* __restrict__ adst, float* __restrict__ as_,
                           float* __restrict__ ad_, int n, int H, int F) {
    int i = blockIdx.x * blockDim.x + threadIdx.x;
    if (i >= n) return;
    int h = i % H;
    const float* xp = xs + (size_t)i * F;       // (node*H + h)*F
    const float* ws = asrc + h * F;
    const float* wd = adst + h * F;
    float s = 0.f, d = 0.f;
    for (int f = 0; f < F; f++) { float v = xp[f]; s += v * ws[f]; d += v * wd[f]; }
    as_[i] = s; ad_[i] = d;
}

__global__ void k_ce(const float* We, const float* ae, float* ce, int H, int F) {
    int h = threadIdx.x;
    if (h >= H) return;
    float s = 0.f;
    for (int f = 0; f < F; f++) s += We[h * F + f] * ae[h * F + f];
    ce[h] = s;
}

__global__ void k_edge_deg(const long long* __restrict__ ei, const float* __restrict__ attr,
                           float* deg, float* suma, int E) {
    int e = blockIdx.x * blockDim.x + threadIdx.x;
    if (e >= E) return;
    int s = (int)ei[e], d = (int)ei[(size_t)E + e];
    if (s != d) { atomicAdd(&deg[d], 1.f); atomicAdd(&suma[d], attr[e]); }
}

__global__ void k_loop_attr(const float* deg, const float* suma, float* loop, int n) {
    int i = blockIdx.x * blockDim.x + threadIdx.x;
    if (i < n) loop[i] = suma[i] / fmaxf(deg[i], 1.f);
}

__global__ void k_self(const float* as_, const float* ad_, const float* loop,
                       const float* ce, float* aself, unsigned* m, int n, int H) {
    int i = blockIdx.x * blockDim.x + threadIdx.x;
    if (i >= n) return;
    int node = i / H, h = i % H;
    float v = lrelu02(as_[i] + ad_[i] + loop[node] * ce[h]);
    aself[i] = v;
    m[i] = f2ord(v);
}

__global__ void k_alpha(const long long* __restrict__ ei, const float* __restrict__ attr,
                        const float* __restrict__ as_, const float* __restrict__ ad_,
                        const float* __restrict__ ce, float* alpha, int E, int H) {
    int i = blockIdx.x * blockDim.x + threadIdx.x;
    if (i >= E * H) return;
    int e = i / H, h = i % H;
    int s = (int)ei[e], d = (int)ei[(size_t)E + e];
    alpha[i] = (s != d) ? lrelu02(as_[s * H + h] + ad_[d * H + h] + attr[e] * ce[h])
                        : -1e9f;
}

__global__ void k_maxe(const long long* __restrict__ ei, const float* __restrict__ alpha,
                       unsigned* m, int E, int H) {
    int i = blockIdx.x * blockDim.x + threadIdx.x;
    if (i >= E * H) return;
    int e = i / H, h = i % H;
    int s = (int)ei[e], d = (int)ei[(size_t)E + e];
    if (s != d) atomicMax(&m[d * H + h], f2ord(alpha[i]));
}

__global__ void k_nodeexp(const unsigned* m, float* aself_ws, float* den, int n) {
    int i = blockIdx.x * blockDim.x + threadIdx.x;
    if (i >= n) return;
    float w = expf(aself_ws[i] - ord2f(m[i]));
    aself_ws[i] = w;    // becomes w_s
    den[i] = w;         // denom init
}

__global__ void k_edgeexp(const long long* __restrict__ ei, const unsigned* __restrict__ m,
                          float* alpha, float* den, int E, int H) {
    int i = blockIdx.x * blockDim.x + threadIdx.x;
    if (i >= E * H) return;
    int e = i / H, h = i % H;
    int s = (int)ei[e], d = (int)ei[(size_t)E + e];
    if (s != d) {
        float w = expf(alpha[i] - ord2f(m[d * H + h]));
        alpha[i] = w;                    // becomes w_e
        atomicAdd(&den[d * H + h], w);
    } else {
        alpha[i] = 0.f;
    }
}

__global__ void k_msginit(const float* ws_, const float* xs, float* msg, int n, int F) {
    int i = blockIdx.x * blockDim.x + threadIdx.x;
    if (i >= n) return;
    msg[i] = ws_[i / F] * xs[i];   // i/F == node*H + h
}

__global__ void k_msgedge(const long long* __restrict__ ei, const float* __restrict__ w_e,
                          const float* __restrict__ xs, float* __restrict__ msg,
                          int E, int H, int F) {
    int i = blockIdx.x * blockDim.x + threadIdx.x;
    if (i >= E * H) return;
    float w = w_e[i];
    if (w == 0.f) return;
    int e = i / H, h = i % H;
    int s = (int)ei[e], d = (int)ei[(size_t)E + e];
    const float* xp = xs + ((size_t)s * H + h) * F;
    float* mp = msg + ((size_t)d * H + h) * F;
    for (int f = 0; f < F; f += 4) {
        float4 v = *(const float4*)&xp[f];
        atomicAdd(&mp[f + 0], w * v.x);
        atomicAdd(&mp[f + 1], w * v.y);
        atomicAdd(&mp[f + 2], w * v.z);
        atomicAdd(&mp[f + 3], w * v.w);
    }
}

// out = elu( bn( msg/den + bias ) + res );  res is read from & written to inout
__global__ void k_combine(const float* __restrict__ msg, const float* __restrict__ den,
                          const float* __restrict__ bias, const float* __restrict__ gamma,
                          const float* __restrict__ beta, const float* __restrict__ mean,
                          const float* __restrict__ var, float* inout, int n, int F, int Co) {
    int i = blockIdx.x * blockDim.x + threadIdx.x;
    if (i >= n) return;
    int c = i % Co;
    float v = msg[i] / den[i / F] + bias[c];
    v = (v - mean[c]) * rsqrtf(var[c] + 1e-5f) * gamma[c] + beta[c] + inout[i];
    inout[i] = v > 0.f ? v : (expf(v) - 1.f);
}

__global__ void k_pool(const float* __restrict__ h3, float* pool, int n) {
    int i = blockIdx.x * blockDim.x + threadIdx.x;
    if (i >= n) return;
    int g = i >> 6, c = i & 63;
    const float* p = h3 + (size_t)g * NPG * 64 + c;
    float s = 0.f;
    for (int k = 0; k < NPG; k++) s += p[(size_t)k * 64];
    pool[i] = s * (1.f / 64.f);
}

__global__ void k_mlp2(const float* __restrict__ m1, const float* __restrict__ W,
                       const float* __restrict__ b, float* out, int G) {
    int i = blockIdx.x * blockDim.x + threadIdx.x;
    if (i >= G * NCLS) return;
    int g = i / NCLS, j = i % NCLS;
    const float* mp = m1 + (size_t)g * 128;
    float s = b[j];
    for (int c = 0; c < 128; c++) {
        float v = mp[c];
        v = v > 0.f ? v : 0.f;          // ReLU fused here
        s += v * W[c * NCLS + j];
    }
    out[i] = s;
}

// ---------------- host-side layer driver ----------------
static inline int cdiv(long long a, int b) { return (int)((a + b - 1) / b); }

static void run_gat_layer(hipStream_t st, const float* act_in, int Ci, int H, int F,
                          const float* Wgat, const float* We, const float* asrc,
                          const float* adst, const float* aedge, const float* gbias,
                          const float* Wres, const float* bres,
                          const float* gamma, const float* beta, const float* mean,
                          const float* var, const long long* ei, const float* eattr,
                          float* XS, float* MSG, float* ALPHA, _Float16* F16,
                          _Float16* W16, float* CE, float* AS, float* AD,
                          float* ASELF, unsigned* M, float* DEN, const float* LOOP,
                          float* act_out) {
    const int Co = H * F, TB = 256;
    k_transpose_f16<<<cdiv((long long)Ci * Co, TB), TB, 0, st>>>(Wgat, W16, Ci, Co);
    k_f32_to_f16<<<cdiv((long long)Nn * Ci, TB), TB, 0, st>>>(act_in, F16, Nn * Ci);
    dim3 gg(Nn / 64, Co / 64);
    k_gemm_wmma<<<gg, 128, 0, st>>>(F16, W16, nullptr, XS, Nn, Ci, Co);
    k_att_pair<<<cdiv((long long)Nn * H, TB), TB, 0, st>>>(XS, asrc, adst, AS, AD, Nn * H, H, F);
    k_ce<<<1, 32, 0, st>>>(We, aedge, CE, H, F);
    k_self<<<cdiv((long long)Nn * H, TB), TB, 0, st>>>(AS, AD, LOOP, CE, ASELF, M, Nn * H, H);
    k_alpha<<<cdiv((long long)En * H, TB), TB, 0, st>>>(ei, eattr, AS, AD, CE, ALPHA, En, H);
    k_maxe<<<cdiv((long long)En * H, TB), TB, 0, st>>>(ei, ALPHA, M, En, H);
    k_nodeexp<<<cdiv((long long)Nn * H, TB), TB, 0, st>>>(M, ASELF, DEN, Nn * H);
    k_edgeexp<<<cdiv((long long)En * H, TB), TB, 0, st>>>(ei, M, ALPHA, DEN, En, H);
    k_msginit<<<cdiv((long long)Nn * Co, TB), TB, 0, st>>>(ASELF, XS, MSG, Nn * Co, F);
    k_msgedge<<<cdiv((long long)En * H, TB), TB, 0, st>>>(ei, ALPHA, XS, MSG, En, H, F);
    // residual linear (same input, already converted in F16)
    k_transpose_f16<<<cdiv((long long)Ci * Co, TB), TB, 0, st>>>(Wres, W16, Ci, Co);
    k_gemm_wmma<<<gg, 128, 0, st>>>(F16, W16, bres, act_out, Nn, Ci, Co);
    k_combine<<<cdiv((long long)Nn * Co, TB), TB, 0, st>>>(MSG, DEN, gbias, gamma, beta,
                                                          mean, var, act_out, Nn * Co, F, Co);
}

extern "C" void kernel_launch(void* const* d_in, const int* in_sizes, int n_in,
                              void* d_out, int out_size, void* d_ws, size_t ws_size,
                              hipStream_t stream) {
    (void)in_sizes; (void)n_in; (void)out_size; (void)ws_size;
    const float*     x     = (const float*)d_in[0];
    const long long* ei    = (const long long*)d_in[1];   // [2,E] int64
    const float*     eattr = (const float*)d_in[2];       // [E,1]
    const float* P[44];
    for (int i = 0; i < 44; i++) P[i] = (const float*)d_in[i];

    char* ws = (char*)d_ws;
    size_t off = 0;
    auto AL = [&](size_t b) { size_t o = off; off += (b + 255) & ~(size_t)255; return ws + o; };

    float*     ACT0  = (float*)AL((size_t)Nn * 256 * 4);
    float*     ACT1  = (float*)AL((size_t)Nn * 256 * 4);
    float*     XS    = (float*)AL((size_t)Nn * 256 * 4);
    float*     MSG   = (float*)AL((size_t)Nn * 256 * 4);
    _Float16*  F16   = (_Float16*)AL((size_t)Nn * 256 * 2);
    float*     ALPHA = (float*)AL((size_t)En * 4 * 4);
    float*     AS    = (float*)AL((size_t)Nn * 4 * 4);
    float*     AD    = (float*)AL((size_t)Nn * 4 * 4);
    float*     ASELF = (float*)AL((size_t)Nn * 4 * 4);
    unsigned*  Mbuf  = (unsigned*)AL((size_t)Nn * 4 * 4);
    float*     DEN   = (float*)AL((size_t)Nn * 4 * 4);
    float*     DEG   = (float*)AL((size_t)Nn * 4);
    float*     SUMA  = (float*)AL((size_t)Nn * 4);
    float*     LOOP  = (float*)AL((size_t)Nn * 4);
    _Float16*  W16   = (_Float16*)AL((size_t)256 * 256 * 2);
    float*     CE    = (float*)AL(256);
    float*     POOL  = (float*)AL((size_t)Gn * 64 * 4);
    _Float16*  P16   = (_Float16*)AL((size_t)Gn * 64 * 2);
    float*     M1    = (float*)AL((size_t)Gn * 128 * 4);

    const int TB = 256;
    // self-loop edge attr (shared by all 3 layers)
    k_zero<<<cdiv(Nn, TB), TB, 0, stream>>>(DEG, Nn);
    k_zero<<<cdiv(Nn, TB), TB, 0, stream>>>(SUMA, Nn);
    k_edge_deg<<<cdiv(En, TB), TB, 0, stream>>>(ei, eattr, DEG, SUMA, En);
    k_loop_attr<<<cdiv(Nn, TB), TB, 0, stream>>>(DEG, SUMA, LOOP, Nn);

    // layer 1: x[.,128] -> ACT0[.,256]
    run_gat_layer(stream, x, INC, HEADS, HIDC,
                  P[4], P[5], P[6], P[7], P[8], P[9],        // g1
                  P[22], P[23],                               // res1
                  P[28], P[29], P[30], P[31],                 // bn1
                  ei, eattr, XS, MSG, ALPHA, F16, W16, CE, AS, AD, ASELF, Mbuf, DEN,
                  LOOP, ACT0);
    // layer 2: ACT0[.,256] -> ACT1[.,256]
    run_gat_layer(stream, ACT0, HIDC * HEADS, HEADS, HIDC,
                  P[10], P[11], P[12], P[13], P[14], P[15],   // g2
                  P[24], P[25],                               // res2
                  P[32], P[33], P[34], P[35],                 // bn2
                  ei, eattr, XS, MSG, ALPHA, F16, W16, CE, AS, AD, ASELF, Mbuf, DEN,
                  LOOP, ACT1);
    // layer 3: ACT1[.,256] -> ACT0[.,64]  (H=1)
    run_gat_layer(stream, ACT1, HIDC * HEADS, 1, HIDC,
                  P[16], P[17], P[18], P[19], P[20], P[21],   // g3
                  P[26], P[27],                               // res3
                  P[36], P[37], P[38], P[39],                 // bn3
                  ei, eattr, XS, MSG, ALPHA, F16, W16, CE, AS, AD, ASELF, Mbuf, DEN,
                  LOOP, ACT0);

    // mean pool + MLP head
    k_pool<<<cdiv((long long)Gn * 64, TB), TB, 0, stream>>>(ACT0, POOL, Gn * 64);
    k_f32_to_f16<<<cdiv((long long)Gn * 64, TB), TB, 0, stream>>>(POOL, P16, Gn * 64);
    k_transpose_f16<<<cdiv(64 * 128, TB), TB, 0, stream>>>(P[40], W16, 64, 128);
    dim3 gm(Gn / 64, 128 / 64);
    k_gemm_wmma<<<gm, 128, 0, stream>>>(P16, W16, P[41], M1, Gn, 64, 128);
    k_mlp2<<<cdiv((long long)Gn * NCLS, TB), TB, 0, stream>>>(M1, P[42], P[43],
                                                             (float*)d_out, Gn);
}
